// PairwiseMINE_67405216743666
// MI455X (gfx1250) — matrix operand order repaired and checked by reading, not compiled, not run on hardware
//
#include <hip/hip_runtime.h>
#include <hip/hip_bf16.h>

typedef __attribute__((ext_vector_type(16))) _Float16 v16h;
typedef __attribute__((ext_vector_type(8)))  float    v8f;

#define B_ROWS 131072
#define D_DIM  64
#define P_PAIRS 32
#define H_DIM  64
#define CHUNKS 32
#define WAVES_PER_BLOCK 8
#define WAVES_PER_PAIR (CHUNKS * WAVES_PER_BLOCK)          /* 256 */
#define ROWS_PER_TILE 8                                    /* 8 joint + 8 marg */
#define TILES_PER_PAIR (B_ROWS / ROWS_PER_TILE)            /* 16384 */
#define TILES_PER_WAVE (TILES_PER_PAIR / WAVES_PER_PAIR)   /* 64 */

#define LOG2E 1.44269504f
#define LN2   0.69314718f
#define HUGEF 3.0e38f

__device__ __forceinline__ v16h splat16(_Float16 x) {
    v16h v;
#pragma unroll
    for (int t = 0; t < 16; ++t) v[t] = x;
    return v;
}

// Weights pre-scaled by log2e, so pl = log2e * preact.
// selu(pre) = max(pre,0) + exp(min(pre,0)) = max(pl,0)*ln2 + exp2(min(pl,0))
__device__ __forceinline__ v16h selu_scaled16(v16h pl, v16h z16, v16h ln2h) {
    v16h m = __builtin_elementwise_min(pl, z16);           // v_pk_min_num_f16
    v16h e;
#pragma unroll
    for (int t = 0; t < 16; ++t)
        e[t] = __builtin_elementwise_exp2(m[t]);           // v_exp_f16
    return __builtin_elementwise_max(pl, z16) * ln2h + e;  // pk_max + pk_fma
}

__global__ __launch_bounds__(256) void PairwiseMINE_main_kernel(
    const float* __restrict__ z,  const float* __restrict__ W1,
    const float* __restrict__ b1, const float* __restrict__ W2,
    const float* __restrict__ b2, const float* __restrict__ W3,
    const float* __restrict__ b3, const int* __restrict__ pairs,
    const int* __restrict__ perms, float* __restrict__ ws)
{
    const int lane = threadIdx.x & 31;
    const int wave = threadIdx.x >> 5;
    const int lm   = lane & 15;          // A row / C column within 16
    const int c    = (lane >> 4) << 3;   // A-fragment K chunk offset: 0 or 8
    const int g16  = (lane >> 4) << 4;   // unused; B fragments take K offset c, as A
    const int p    = blockIdx.y;
    const int waveInPair = blockIdx.x * WAVES_PER_BLOCK + wave;
    const int isMarg = (lm >> 3) & 1;    // A rows 8..15 carry shuffled z_i

    const int pi = pairs[2 * p];
    const int pj = pairs[2 * p + 1];

    // ---- W2 B-fragments, pre-scaled by log2e ----
    v16h bfrag[4][2];
#pragma unroll
    for (int n = 0; n < 4; ++n)
#pragma unroll
        for (int kc = 0; kc < 2; ++kc)
#pragma unroll
            for (int t = 0; t < 16; ++t) {
                int K = 32 * kc + ((t < 8) ? (c + t) : (8 + c + t));
                bfrag[n][kc][t] = (_Float16)(W2[K * H_DIM + 16 * n + lm] * LOG2E);
            }

    // ---- layer-1 weights (A-fragment half order), pre-scaled by log2e ----
    v16h w10a, w10b, w11a, w11b, b1a, b1b;
#pragma unroll
    for (int t = 0; t < 16; ++t) {
        int jl = (t < 8) ? (c + t) : (8 + c + t);   // K for a_lo (0..31)
        int jh = jl + 32;                            // K for a_hi (32..63)
        w10a[t] = (_Float16)(W1[jl] * LOG2E);
        w10b[t] = (_Float16)(W1[jh] * LOG2E);
        w11a[t] = (_Float16)(W1[H_DIM + jl] * LOG2E);
        w11b[t] = (_Float16)(W1[H_DIM + jh] * LOG2E);
        b1a[t]  = (_Float16)(b1[jl] * LOG2E);
        b1b[t]  = (_Float16)(b1[jh] * LOG2E);
    }

    // ---- per-column constants (exp2-domain readout) ----
    float w3v[4], w3e[4], cseed[4];
#pragma unroll
    for (int n = 0; n < 4; ++n) {
        w3v[n] = W3[16 * n + lm];
        w3e[n] = w3v[n] * LOG2E;
        float cs = 0.f;      // colsum of scaled W2 column (this lane: 32 of 64 K)
#pragma unroll
        for (int kc = 0; kc < 2; ++kc)
#pragma unroll
            for (int t = 0; t < 16; ++t) cs += (float)bfrag[n][kc][t];
        cs += __shfl_xor(cs, 16, 32);
        cseed[n] = b2[16 * n + lm] * LOG2E - cs;  // absorbs layer-1 "-1" shift
    }
    float s3 = w3v[0] + w3v[1] + w3v[2] + w3v[3];
    s3 += __shfl_xor(s3, 1, 32);
    s3 += __shfl_xor(s3, 2, 32);
    s3 += __shfl_xor(s3, 4, 32);
    s3 += __shfl_xor(s3, 8, 32);
    const float badj = b3[0] - s3;       // true score = ln2*sum(tk') + badj
    // half-wave select masks (C rows 0-7 -> lanes 0-15 joint; 8-15 -> 16-31 marg)
    const float mJln = (lane < 16) ? LN2 : 0.f;
    const float mME  = (lane < 16) ? 0.f : __expf(badj);

    const v16h z16  = splat16((_Float16)0.0f);
    const v16h ln2h = splat16((_Float16)LN2);

    float accJ = 0.f, accE = 0.f;
    for (int it = 0; it < TILES_PER_WAVE; ++it) {
        int tile = waveInPair + it * WAVES_PER_PAIR;
        unsigned r = (unsigned)tile * ROWS_PER_TILE + (lm & 7);
        int   pr  = perms[(size_t)p * B_ROWS + r];
        unsigned r0 = isMarg ? (unsigned)pr : r;         // shuffled z_i rows
        float x1 = z[(size_t)r * D_DIM + pj];
        float x0 = z[(size_t)r0 * D_DIM + pi];

        // ---- layer 1 -> A fragments (rows 0-7 joint, 8-15 marg) ----
        v16h xv0 = splat16((_Float16)x0);
        v16h xv1 = splat16((_Float16)x1);
        v16h alo = selu_scaled16(xv0 * w10a + (xv1 * w11a + b1a), z16, ln2h);
        v16h ahi = selu_scaled16(xv0 * w10b + (xv1 * w11b + b1b), z16, ln2h);

        // ---- layer 2: kc-outer schedule -> 4 independent WMMAs per group ----
        v8f cf[4];
#pragma unroll
        for (int n = 0; n < 4; ++n)
#pragma unroll
            for (int k = 0; k < 8; ++k) cf[n][k] = cseed[n];
#pragma unroll
        for (int n = 0; n < 4; ++n)
            cf[n] = __builtin_amdgcn_wmma_f32_16x16x32_f16(
                        false, alo, false, bfrag[n][0], (short)0, cf[n], false, false);
#pragma unroll
        for (int n = 0; n < 4; ++n)
            cf[n] = __builtin_amdgcn_wmma_f32_16x16x32_f16(
                        false, ahi, false, bfrag[n][1], (short)0, cf[n], false, false);

        // ---- epilogue: tk in exp2 domain; med3 = select-free min/max ----
        float tk[8];
#pragma unroll
        for (int k = 0; k < 8; ++k) tk[k] = 0.f;
#pragma unroll
        for (int n = 0; n < 4; ++n)
#pragma unroll
            for (int k = 0; k < 8; ++k) {
                float x  = cf[n][k];
                float mn = __builtin_amdgcn_fmed3f(x, -HUGEF, 0.0f);  // min(x,0)
                float mx = __builtin_amdgcn_fmed3f(x, 0.0f,  HUGEF);  // max(x,0)
                float e  = __builtin_amdgcn_exp2f(mn);
                tk[k] = fmaf(mx, w3v[n], fmaf(e, w3e[n], tk[k]));
            }

        // joint half: linear readout, per-lane partial (true score*log2e)
        float sumk = 0.f;
#pragma unroll
        for (int k = 0; k < 8; ++k) sumk += tk[k];
        accJ = fmaf(sumk, mJln, accJ);

        // marg half: per-row score via 16-lane tree, then exp2
        float esum = 0.f;
#pragma unroll
        for (int k = 0; k < 8; ++k) {
            float s = tk[k];
            s += __shfl_xor(s, 1, 32);
            s += __shfl_xor(s, 2, 32);
            s += __shfl_xor(s, 4, 32);
            s += __shfl_xor(s, 8, 32);
            esum += __builtin_amdgcn_exp2f(s);   // exp(score-badj); badj in mME
        }
        accE = fmaf(esum, mME, accE);
    }

    // wave totals: accJ partials unique (lanes 0-15); accE rows replicated 16x
    accJ += __shfl_xor(accJ, 16, 32);
    accJ += __shfl_xor(accJ,  8, 32);
    accJ += __shfl_xor(accJ,  4, 32);
    accJ += __shfl_xor(accJ,  2, 32);
    accJ += __shfl_xor(accJ,  1, 32);
    accE += __shfl_xor(accE, 16, 32);
    accE += __shfl_xor(accE,  8, 32);
    accE += __shfl_xor(accE,  4, 32);
    accE += __shfl_xor(accE,  2, 32);
    accE += __shfl_xor(accE,  1, 32);

    if (lane == 0) {
        size_t slot = ((size_t)p * WAVES_PER_PAIR + waveInPair) * 2;
        ws[slot + 0] = accJ;                  // sum of (score - badj) over rows
        ws[slot + 1] = accE * (1.f / 16.f);   // sum of exp(score) over rows
    }
}

__global__ __launch_bounds__(32) void PairwiseMINE_reduce_kernel(
    const float* __restrict__ ws, const float* __restrict__ W3,
    const float* __restrict__ b3, float* __restrict__ out)
{
    int p = threadIdx.x;              // one lane per pair (P == 32 == wave32)
    float sj = 0.f, se = 0.f;
    for (int w = 0; w < WAVES_PER_PAIR; ++w) {
        sj += ws[((size_t)p * WAVES_PER_PAIR + w) * 2 + 0];
        se += ws[((size_t)p * WAVES_PER_PAIR + w) * 2 + 1];
    }
    float s3 = 0.f;
    for (int j = 0; j < H_DIM; ++j) s3 += W3[j];
    float badj = b3[0] - s3;          // constant omitted from joint partials
    float mi = (sj * (1.f / B_ROWS) + badj) - __logf(se * (1.f / B_ROWS));
    mi += __shfl_xor(mi, 16, 32);
    mi += __shfl_xor(mi,  8, 32);
    mi += __shfl_xor(mi,  4, 32);
    mi += __shfl_xor(mi,  2, 32);
    mi += __shfl_xor(mi,  1, 32);
    if (p == 0) out[0] = mi * (1.f / P_PAIRS);
}

extern "C" void kernel_launch(void* const* d_in, const int* in_sizes, int n_in,
                              void* d_out, int out_size, void* d_ws, size_t ws_size,
                              hipStream_t stream) {
    const float* z     = (const float*)d_in[0];
    const float* W1    = (const float*)d_in[1];
    const float* b1    = (const float*)d_in[2];
    const float* W2    = (const float*)d_in[3];
    const float* b2    = (const float*)d_in[4];
    const float* W3    = (const float*)d_in[5];
    const float* b3    = (const float*)d_in[6];
    const int*   pairs = (const int*)d_in[7];
    const int*   perms = (const int*)d_in[8];
    float* out = (float*)d_out;
    float* ws  = (float*)d_ws;   // P * 256 waves * 2 floats = 64 KB

    dim3 grid(CHUNKS, P_PAIRS, 1);
    PairwiseMINE_main_kernel<<<grid, dim3(256, 1, 1), 0, stream>>>(
        z, W1, b1, W2, b2, W3, b3, pairs, perms, ws);
    PairwiseMINE_reduce_kernel<<<1, 32, 0, stream>>>(ws, W3, b3, out);
}
